// BertAttention_6734508720438
// MI455X (gfx1250) — matrix operand order repaired and checked by reading, not compiled, not run on hardware
//
#include <hip/hip_runtime.h>
#include <hip/hip_bf16.h>

// ---------------------------------------------------------------------------
// BERT attention layer for MI455X (gfx1250, wave32, WMMA).
//   q   = hs @ Wq^T + bq                (WMMA bf16 GEMM, bf16 out in [B,H,S,D])
//   att = flash-attention(q,q,q,mask)   (WMMA bf16, online softmax, LDS tiles)
//   y   = ctx @ Wo^T + bo               (WMMA bf16 GEMM, fp32 out)
//   out = LayerNorm(y + hs)             (wave32 reductions)
// ---------------------------------------------------------------------------

#define BB   8
#define SS   1024
#define HIDD 1024
#define NHH  16
#define HDD  64

typedef __attribute__((ext_vector_type(16))) __bf16 v16bf;
typedef __attribute__((ext_vector_type(8)))  __bf16 v8bf;
typedef __attribute__((ext_vector_type(8)))  float  v8f;

union Frag16 { v16bf v; v8bf h[2]; __bf16 e[16]; };

// convert 8 contiguous fp32 -> 8 bf16 (two dwordx4 global loads)
__device__ __forceinline__ void cvt8_bf16(const float* __restrict__ p, __bf16* d) {
  float4 a = *reinterpret_cast<const float4*>(p);
  float4 b = *reinterpret_cast<const float4*>(p + 4);
  d[0] = (__bf16)a.x; d[1] = (__bf16)a.y; d[2] = (__bf16)a.z; d[3] = (__bf16)a.w;
  d[4] = (__bf16)b.x; d[5] = (__bf16)b.y; d[6] = (__bf16)b.z; d[7] = (__bf16)b.w;
}

// ---------------------------------------------------------------------------
// GEMM: out[m,n] = sum_k A[m,k] * W[n,k] + bias[n]        (M x 1024 x 1024)
// Register-blocked: each wave owns a 16x64 tile (4 accumulators), so the A
// fragment is reused 4x per k-step. K unrolled by 2 -> 8 WMMAs per loop body.
// mode 0: emit bf16 into q buffer laid out [B, NH, S, HD]
// mode 1: emit fp32 into out_f32 [M, 1024]
// ---------------------------------------------------------------------------
__global__ void __launch_bounds__(256)
gemm_bf16_wmma(const float* __restrict__ A, const float* __restrict__ W,
               const float* __restrict__ bias, float* __restrict__ out_f32,
               __bf16* __restrict__ q_out, int mode) {
  const int lane = threadIdx.x & 31;
  const int wid  = threadIdx.x >> 5;
  const int half = lane >> 4;
  const int l15  = lane & 15;
  const int ntb  = (blockIdx.x * 8 + wid) * 4;  // 4 consecutive n-tiles (N=1024)
  const int mtile = blockIdx.y;                 // 0..511 (M = 8192)
  const int row  = mtile * 16 + l15;            // A row this lane feeds (A frag)
  const float* arow = A + (size_t)row * HIDD;
  const float* wrow[4];
#pragma unroll
  for (int t = 0; t < 4; ++t)
    wrow[t] = W + (size_t)((ntb + t) * 16 + l15) * HIDD;   // B frag: lane = col

  v8f acc[4] = {};
  for (int k0 = 0; k0 < HIDD; k0 += 64) {
#pragma unroll
    for (int kk = 0; kk < 64; kk += 32) {
      const int k = k0 + kk;
      Frag16 a;
      // A 16x32 layout: lanes 0-15 K=[0..15], lanes 16-31 K=[16..31]
      cvt8_bf16(arow + k + 16 * half,     &a.e[0]);
      cvt8_bf16(arow + k + 16 * half + 8, &a.e[8]);
#pragma unroll
      for (int t = 0; t < 4; ++t) {
        Frag16 b;  // B 32x16 layout: lane=col, K = e + 16*half
        cvt8_bf16(wrow[t] + k + 16 * half,     &b.e[0]);
        cvt8_bf16(wrow[t] + k + 16 * half + 8, &b.e[8]);
        acc[t] = __builtin_amdgcn_wmma_f32_16x16x32_bf16(false, a.v, false, b.v,
                                                         (short)0, acc[t], false, false);
      }
    }
  }

#pragma unroll
  for (int t = 0; t < 4; ++t) {
    const int col = (ntb + t) * 16 + l15;
    const float bn = bias[col];
    if (mode == 0) {
      const int h = col >> 6, d = col & 63;
#pragma unroll
      for (int v = 0; v < 8; ++v) {
        const int m = mtile * 16 + v + 8 * half;           // global row
        const int bidx = m >> 10, s = m & 1023;
        q_out[(((size_t)(bidx * NHH + h)) * SS + s) * HDD + d] =
            (__bf16)(acc[t][v] + bn);
      }
    } else {
#pragma unroll
      for (int v = 0; v < 8; ++v) {
        const int m = mtile * 16 + v + 8 * half;
        out_f32[(size_t)m * HIDD + col] = acc[t][v] + bn;
      }
    }
  }
}

// ---------------------------------------------------------------------------
// Flash attention (Q = K = V). Workgroup = one (b,h), 8 waves x 16 query rows
// = 128 rows per block. Keys streamed in blocks of 32 through LDS (row-major
// + transposed copies). Online softmax; 8 WMMAs per key-block per wave.
// ---------------------------------------------------------------------------
__global__ void __launch_bounds__(256)
attn_flash_wmma(const __bf16* __restrict__ qbh,     // [B*NH, S, 64] bf16
                const float* __restrict__ mask,     // [B, S]
                float* __restrict__ ctx) {          // [B, S, 1024] fp32
  __shared__ __bf16 kb[32 * 64];     // K block  [key][d]
  __shared__ __bf16 kbT[64 * 32];    // K block^T [d][key]
  __shared__ __bf16 pbuf[8][16 * 32];// per-wave prob scratch [m][key]

  const int lane = threadIdx.x & 31;
  const int wid  = threadIdx.x >> 5;
  const int half = lane >> 4;
  const int l15  = lane & 15;
  const int bh   = blockIdx.y;               // 0..127
  const int rb   = blockIdx.x;               // 0..7 row-blocks of 128
  const int b    = bh >> 4, h = bh & 15;
  const int srow = rb * 128 + wid * 16;      // first query row of this wave

  // Q fragments for this wave's 16 rows (contraction d: [0..31], [32..63])
  Frag16 aq0, aq1;
  {
    const __bf16* qr = qbh + ((size_t)bh * SS + srow + l15) * HDD;
    aq0.h[0] = *reinterpret_cast<const v8bf*>(qr + 16 * half);
    aq0.h[1] = *reinterpret_cast<const v8bf*>(qr + 16 * half + 8);
    aq1.h[0] = *reinterpret_cast<const v8bf*>(qr + 32 + 16 * half);
    aq1.h[1] = *reinterpret_cast<const v8bf*>(qr + 40 + 16 * half);
  }

  v8f accO[4] = {};                 // 16x64 context accumulator (4 N-tiles)
  float mrun[8], lrun[8];
#pragma unroll
  for (int v = 0; v < 8; ++v) { mrun[v] = -1e30f; lrun[v] = 0.f; }

  for (int kb0 = 0; kb0 < SS; kb0 += 32) {
    __syncthreads();
    { // cooperative load: 32 keys x 64 dims, also transposed copy
      const int key = threadIdx.x >> 3;
      const int db  = (threadIdx.x & 7) * 8;
      v8bf qv = *reinterpret_cast<const v8bf*>(
          qbh + ((size_t)bh * SS + kb0 + key) * HDD + db);
      *reinterpret_cast<v8bf*>(&kb[key * 64 + db]) = qv;
#pragma unroll
      for (int i = 0; i < 8; ++i) kbT[(db + i) * 32 + key] = qv[i];
    }
    __syncthreads();

    // ---- scores: S = Q (16x64) @ K^T (64x32) -> two 16x16 tiles -----------
    v8f sc0 = {}, sc1 = {};
#pragma unroll
    for (int dstep = 0; dstep < 2; ++dstep) {
      Frag16 bk0, bk1;   // B frag: lane=key col, k = d = dstep*32 + e + 16*half
      const int doff = dstep * 32 + 16 * half;
      bk0.h[0] = *reinterpret_cast<const v8bf*>(&kb[(l15)      * 64 + doff]);
      bk0.h[1] = *reinterpret_cast<const v8bf*>(&kb[(l15)      * 64 + doff + 8]);
      bk1.h[0] = *reinterpret_cast<const v8bf*>(&kb[(16 + l15) * 64 + doff]);
      bk1.h[1] = *reinterpret_cast<const v8bf*>(&kb[(16 + l15) * 64 + doff + 8]);
      const v16bf aq = dstep ? aq1.v : aq0.v;
      sc0 = __builtin_amdgcn_wmma_f32_16x16x32_bf16(false, aq, false, bk0.v,
                                                    (short)0, sc0, false, false);
      sc1 = __builtin_amdgcn_wmma_f32_16x16x32_bf16(false, aq, false, bk1.v,
                                                    (short)0, sc1, false, false);
    }

    // ---- scale + mask + online softmax ------------------------------------
    const float mk0 = mask[b * SS + kb0 + l15];
    const float mk1 = mask[b * SS + kb0 + 16 + l15];
#pragma unroll
    for (int v = 0; v < 8; ++v) {
      float s0 = sc0[v] * 0.125f + mk0;      // 1/sqrt(64)
      float s1 = sc1[v] * 0.125f + mk1;
      float mv = fmaxf(s0, s1);
#pragma unroll
      for (int off = 1; off < 16; off <<= 1) mv = fmaxf(mv, __shfl_xor(mv, off));
      const float mnew = fmaxf(mrun[v], mv);
      const float corr = __expf(mrun[v] - mnew);
      mrun[v] = mnew;
      s0 = __expf(s0 - mnew);
      s1 = __expf(s1 - mnew);
      float rs = s0 + s1;
#pragma unroll
      for (int off = 1; off < 16; off <<= 1) rs += __shfl_xor(rs, off);
      lrun[v] = lrun[v] * corr + rs;
#pragma unroll
      for (int t = 0; t < 4; ++t) accO[t][v] *= corr;
      sc0[v] = s0; sc1[v] = s1;
    }

    // ---- transpose P (C/D layout) -> A-fragment via wave-private LDS ------
    __bf16* pw = &pbuf[wid][0];
#pragma unroll
    for (int v = 0; v < 8; ++v) {
      const int m = v + 8 * half;
      pw[m * 32 + l15]      = (__bf16)sc0[v];
      pw[m * 32 + 16 + l15] = (__bf16)sc1[v];
    }
    Frag16 pa;  // DS ops are in-order within a wave; compiler waits DScnt
    pa.h[0] = *reinterpret_cast<const v8bf*>(&pw[l15 * 32 + 16 * half]);
    pa.h[1] = *reinterpret_cast<const v8bf*>(&pw[l15 * 32 + 16 * half + 8]);

    // ---- accO += P (16x32) @ V (32x64), 4 N-tiles of 16 -------------------
#pragma unroll
    for (int t = 0; t < 4; ++t) {
      Frag16 vb;  // B frag: lane = d col, k = key = e + 16*half
      const int d = t * 16 + l15;
      vb.h[0] = *reinterpret_cast<const v8bf*>(&kbT[d * 32 + 16 * half]);
      vb.h[1] = *reinterpret_cast<const v8bf*>(&kbT[d * 32 + 16 * half + 8]);
      accO[t] = __builtin_amdgcn_wmma_f32_16x16x32_bf16(false, pa.v, false, vb.v,
                                                        (short)0, accO[t], false, false);
    }
  }

  // ---- normalize and scatter into ctx [B,S,HID] (undoes head transpose) ---
#pragma unroll
  for (int t = 0; t < 4; ++t) {
    const int d = t * 16 + l15;
#pragma unroll
    for (int v = 0; v < 8; ++v) {
      const int s = srow + v + 8 * half;
      ctx[((size_t)(b * SS + s)) * HIDD + h * HDD + d] = accO[t][v] / lrun[v];
    }
  }
}

// ---------------------------------------------------------------------------
// out = LayerNorm(y + hs) over rows of 1024; one block per row.
// ---------------------------------------------------------------------------
__global__ void __launch_bounds__(256)
residual_layernorm(const float* __restrict__ y, const float* __restrict__ hs,
                   const float* __restrict__ w, const float* __restrict__ bias,
                   float* __restrict__ out) {
  const int row = blockIdx.x;
  const int lane = threadIdx.x & 31;
  const int wid  = threadIdx.x >> 5;
  const float* yr = y  + (size_t)row * HIDD;
  const float* hr = hs + (size_t)row * HIDD;

  float x[4], s1 = 0.f, s2 = 0.f;
#pragma unroll
  for (int i = 0; i < 4; ++i) {
    const int c = threadIdx.x + i * 256;
    x[i] = yr[c] + hr[c];
    s1 += x[i];
    s2 += x[i] * x[i];
  }
#pragma unroll
  for (int off = 16; off; off >>= 1) {
    s1 += __shfl_xor(s1, off);
    s2 += __shfl_xor(s2, off);
  }
  __shared__ float sh1[8], sh2[8];
  if (lane == 0) { sh1[wid] = s1; sh2[wid] = s2; }
  __syncthreads();
  if (wid == 0) {
    float a = (lane < 8) ? sh1[lane] : 0.f;
    float c = (lane < 8) ? sh2[lane] : 0.f;
#pragma unroll
    for (int off = 4; off; off >>= 1) { a += __shfl_xor(a, off); c += __shfl_xor(c, off); }
    if (lane == 0) { sh1[0] = a; sh2[0] = c; }
  }
  __syncthreads();
  const float mean = sh1[0] * (1.f / HIDD);
  const float var  = sh2[0] * (1.f / HIDD) - mean * mean;
  const float inv  = rsqrtf(var + 1e-12f);
#pragma unroll
  for (int i = 0; i < 4; ++i) {
    const int c = threadIdx.x + i * 256;
    out[(size_t)row * HIDD + c] = w[c] * ((x[i] - mean) * inv) + bias[c];
  }
}

// ---------------------------------------------------------------------------
extern "C" void kernel_launch(void* const* d_in, const int* in_sizes, int n_in,
                              void* d_out, int out_size, void* d_ws, size_t ws_size,
                              hipStream_t stream) {
  const float* hs   = (const float*)d_in[0];
  const float* mask = (const float*)d_in[1];
  const float* Wq   = (const float*)d_in[2];
  const float* bq   = (const float*)d_in[3];
  const float* Wo   = (const float*)d_in[4];
  const float* bo   = (const float*)d_in[5];
  const float* lnw  = (const float*)d_in[6];
  const float* lnb  = (const float*)d_in[7];
  float* out = (float*)d_out;

  char* ws = (char*)d_ws;
  __bf16* qb  = (__bf16*)ws;                          // 16 MB: q bf16 [B,H,S,64]
  float*  ctx = (float*)(ws + (16u << 20));           // 32 MB: ctx fp32 [B,S,HID]
  float*  yb  = (float*)(ws + (48u << 20));           // 32 MB: pre-LN fp32

  // 1) q = hs @ Wq^T + bq  (bf16 out, BHSD layout)
  gemm_bf16_wmma<<<dim3(2, 512), 256, 0, stream>>>(hs, Wq, bq, nullptr, qb, 0);
  // 2) flash attention (Q=K=V), ctx in [B,S,HID]
  attn_flash_wmma<<<dim3(8, 128), 256, 0, stream>>>(qb, mask, ctx);
  // 3) y = ctx @ Wo^T + bo  (fp32 out)
  gemm_bf16_wmma<<<dim3(2, 512), 256, 0, stream>>>(ctx, Wo, bo, yb, nullptr, 1);
  // 4) out = LN(y + hs)
  residual_layernorm<<<dim3(BB * SS), 256, 0, stream>>>(yb, hs, lnw, lnb, out);
}
